// GNN3_56667798503740
// MI455X (gfx1250) — matrix operand order repaired and checked by reading, not compiled, not run on hardware
//
#include <hip/hip_runtime.h>

typedef __attribute__((ext_vector_type(16))) _Float16 v16h;
typedef __attribute__((ext_vector_type(8)))  _Float16 v8h;
typedef __attribute__((ext_vector_type(8)))  float    v8f;

#define NN 50000
#define FF 128
#define NHEADS 4
#define CD 32
#define HCD 128
#define KOUT 10
#define NE 800000
#define NET (NE + NN)          // 850000 edges incl. self loops
#define NSLOPE 0.2f
#define GEPS 1e-5f

// ---- monotonic float <-> uint encoding for atomicMax over signed floats ----
__device__ __forceinline__ unsigned fenc(float f) {
  unsigned u = __float_as_uint(f);
  return (u & 0x80000000u) ? ~u : (u | 0x80000000u);
}
__device__ __forceinline__ float fdec(unsigned k) {
  return (k & 0x80000000u) ? __uint_as_float(k & 0x7FFFFFFFu)
                           : __uint_as_float(~k);
}

__global__ void k_set_u32(unsigned* __restrict__ p, unsigned v, int n) {
  int i = blockIdx.x * blockDim.x + threadIdx.x;
  if (i < n) p[i] = v;
}

// ---- GraphNorm: fused column sum / sumsq ----
__global__ void k_col_stats(const float* __restrict__ x, float* __restrict__ sums) {
  int c = threadIdx.x;                      // 0..127, coalesced across columns
  float s = 0.f, ss = 0.f;
  for (int r = blockIdx.x; r < NN; r += gridDim.x) {
    float v = x[r * FF + c];
    s += v; ss += v * v;
  }
  atomicAdd(&sums[c], s);
  atomicAdd(&sums[FF + c], ss);
}

__global__ void k_gn_finalize(const float* __restrict__ sums,
                              const float* __restrict__ gw,
                              const float* __restrict__ gb,
                              const float* __restrict__ gms,
                              float* __restrict__ prm) {
  int c = threadIdx.x;
  float mean = sums[c] / (float)NN;
  float ma = gms[c] * mean;                 // ms * mean
  // var of (x - ms*mean) = E[x^2] - 2*ma*E[x] + ma^2
  float var = sums[FF + c] / (float)NN - 2.f * ma * mean + ma * ma;
  prm[c] = ma;
  prm[FF + c] = rsqrtf(var + GEPS) * gw[c];
  prm[2 * FF + c] = gb[c];
}

__global__ void k_gn_apply(const float* __restrict__ x, const float* __restrict__ prm,
                           _Float16* __restrict__ h16) {
  int i = blockIdx.x * blockDim.x + threadIdx.x;
  if (i >= NN * FF) return;
  int c = i & (FF - 1);
  h16[i] = (_Float16)((x[i] - prm[c]) * prm[FF + c] + prm[2 * FF + c]);
}

// ---- weight [128 x mout] f32 -> transposed [mout x 128] f16 ----
__global__ void k_wt16(const float* __restrict__ w, _Float16* __restrict__ wt, int mout) {
  int i = blockIdx.x * blockDim.x + threadIdx.x;
  if (i >= 128 * mout) return;
  int k = i / mout, m = i - k * mout;
  wt[m * 128 + k] = (_Float16)w[i];
}

// ---- out[NN x mout] = A16[NN x 128] @ W + bias, via v_wmma_f32_16x16x32_f16
// grid.x = NN/16 row tiles, blockDim = 32 * (mout/16), one wave per 16x16 tile
__global__ void k_gemm_wmma(const _Float16* __restrict__ A,
                            const _Float16* __restrict__ Bt,   // [mout x 128]
                            const float* __restrict__ bias,
                            float* __restrict__ out, int mout) {
  int lane = threadIdx.x & 31;
  int wave = threadIdx.x >> 5;
  int row0 = blockIdx.x * 16;
  // A fragment (16x32 f16): lane = row (lane&15), K-half select (lane>>4)
  const _Float16* Arow = A + (row0 + (lane & 15)) * 128;
  int khalf = (lane >> 4) * 8;
  // B fragment (32x16 f16): lane = col (lane&15), K-block select (lane>>4)
  int col = wave * 16 + (lane & 15);
  const _Float16* Bcol = Bt + col * 128;
  int kb = (lane >> 4) * 8;
  v8f acc = {};
#pragma unroll
  for (int kk = 0; kk < 128; kk += 32) {
    v8h a0 = *(const v8h*)(Arow + kk + khalf);        // K = kk+khalf .. +7
    v8h a1 = *(const v8h*)(Arow + kk + 16 + khalf);   // K = kk+16+khalf .. +7
    v8h b0 = *(const v8h*)(Bcol + kk + kb);           // K = kk+kb .. +7
    v8h b1 = *(const v8h*)(Bcol + kk + 16 + kb);      // K = kk+16+kb .. +7
    v16h a, b;
#pragma unroll
    for (int i = 0; i < 8; ++i) { a[i] = a0[i]; a[i + 8] = a1[i];
                                  b[i] = b0[i]; b[i + 8] = b1[i]; }
    acc = __builtin_amdgcn_wmma_f32_16x16x32_f16(
        false, a, false, b, (short)0, acc, false, false);
  }
  float bv = bias[col];
  int rofs = (lane < 16) ? 0 : 8;     // C/D layout: VGPR r -> M=r (lo half) / M=r+8
#pragma unroll
  for (int r = 0; r < 8; ++r)
    out[(row0 + r + rofs) * mout + col] = acc[r] + bv;
}

// ---- edge attention scores + segment max (one wave == one head, C=32) ----
__global__ void k_edge_attn(const float* __restrict__ xl, const float* __restrict__ xr,
                            const int* __restrict__ ei, const float* __restrict__ att,
                            float* __restrict__ ebuf, unsigned* __restrict__ nmax,
                            int heads) {
  int tpe = heads * 32;
  int epb = 256 / tpe;
  int le = threadIdx.x / tpe;
  int li = threadIdx.x % tpe;
  int e = blockIdx.x * epb + le;
  if (e >= NET) return;
  int s, d;
  if (e < NE) { s = ei[e]; d = ei[NE + e]; } else { s = d = e - NE; }
  float v = xl[s * tpe + li] + xr[d * tpe + li];
  v = (v > 0.f) ? v : v * NSLOPE;               // leaky_relu
  float p = v * att[li];
#pragma unroll
  for (int off = 16; off; off >>= 1) p += __shfl_xor(p, off, 32);
  if ((li & 31) == 0) {
    int h = li >> 5;
    ebuf[e * heads + h] = p;
    atomicMax(&nmax[d * heads + h], fenc(p));
  }
}

// ---- exp(e - max[dst]) and segment-sum denominator ----
__global__ void k_edge_exp(const int* __restrict__ ei, float* __restrict__ ebuf,
                           const unsigned* __restrict__ nmax, float* __restrict__ denom,
                           int heads) {
  int t = blockIdx.x * blockDim.x + threadIdx.x;
  if (t >= NET * heads) return;
  int e = t / heads, h = t - e * heads;
  int d = (e < NE) ? ei[NE + e] : (e - NE);
  float ex = expf(ebuf[t] - fdec(nmax[d * heads + h]));
  ebuf[t] = ex;
  atomicAdd(&denom[d * heads + h], ex);
}

// ---- alpha-weighted scatter aggregation ----
__global__ void k_edge_aggr(const float* __restrict__ xl, const int* __restrict__ ei,
                            const float* __restrict__ ebuf, const float* __restrict__ denom,
                            float* __restrict__ agg, int heads) {
  int tpe = heads * 32;
  int epb = 256 / tpe;
  int le = threadIdx.x / tpe;
  int li = threadIdx.x % tpe;
  int e = blockIdx.x * epb + le;
  if (e >= NET) return;
  int s, d;
  if (e < NE) { s = ei[e]; d = ei[NE + e]; } else { s = d = e - NE; }
  int h = li >> 5;
  float alpha = ebuf[e * heads + h] / denom[d * heads + h];
  atomicAdd(&agg[d * tpe + li], alpha * xl[s * tpe + li]);
}

// ---- +bias, ELU, convert to f16 for next layer's GEMM ----
__global__ void k_elu16(const float* __restrict__ agg, const float* __restrict__ bias,
                        _Float16* __restrict__ h16) {
  int i = blockIdx.x * blockDim.x + threadIdx.x;
  if (i >= NN * HCD) return;
  float v = agg[i] + bias[i & (HCD - 1)];
  v = (v > 0.f) ? v : (expf(v) - 1.f);
  h16[i] = (_Float16)v;
}

// ---- final: out = (agg3 + bias3) @ wlin + blin ----
__global__ void k_final(const float* __restrict__ agg3, const float* __restrict__ b3,
                        const float* __restrict__ wl, const float* __restrict__ bl,
                        float* __restrict__ out) {
  int i = blockIdx.x * blockDim.x + threadIdx.x;
  if (i >= NN * KOUT) return;
  int n = i / KOUT, k = i - n * KOUT;
  float s = bl[k];
#pragma unroll
  for (int c = 0; c < CD; ++c) s += (agg3[n * CD + c] + b3[c]) * wl[c * KOUT + k];
  out[i] = s;
}

extern "C" void kernel_launch(void* const* d_in, const int* in_sizes, int n_in,
                              void* d_out, int out_size, void* d_ws, size_t ws_size,
                              hipStream_t stream) {
  const float* x   = (const float*)d_in[0];
  const int*   ei  = (const int*)d_in[1];
  const float* gw  = (const float*)d_in[2];
  const float* gb  = (const float*)d_in[3];
  const float* gms = (const float*)d_in[4];
  const float* W[3][2]  = {{(const float*)d_in[5],  (const float*)d_in[7]},
                           {(const float*)d_in[11], (const float*)d_in[13]},
                           {(const float*)d_in[17], (const float*)d_in[19]}};
  const float* Bv[3][2] = {{(const float*)d_in[6],  (const float*)d_in[8]},
                           {(const float*)d_in[12], (const float*)d_in[14]},
                           {(const float*)d_in[18], (const float*)d_in[20]}};
  const float* att[3]   = {(const float*)d_in[9],  (const float*)d_in[15], (const float*)d_in[21]};
  const float* cbias[3] = {(const float*)d_in[10], (const float*)d_in[16], (const float*)d_in[22]};
  const float* wlin = (const float*)d_in[23];
  const float* blin = (const float*)d_in[24];
  float* out = (float*)d_out;

  // workspace carve-out (256B aligned); total ~105 MB
  char* ws = (char*)d_ws;
  size_t off = 0;
  auto carve = [&](size_t bytes) -> void* {
    void* p = ws + off;
    off = (off + bytes + 255) & ~(size_t)255;
    return p;
  };
  _Float16* h16  = (_Float16*)carve((size_t)NN * HCD * 2);
  float*    xl   = (float*)   carve((size_t)NN * HCD * 4);
  float*    xr   = (float*)   carve((size_t)NN * HCD * 4);
  float*    agg  = (float*)   carve((size_t)NN * HCD * 4);
  float*    ebuf = (float*)   carve((size_t)NET * NHEADS * 4);
  unsigned* nmax = (unsigned*)carve((size_t)NN * NHEADS * 4);
  float*    denom= (float*)   carve((size_t)NN * NHEADS * 4);
  float*    sums = (float*)   carve(2 * FF * 4);
  float*    prm  = (float*)   carve(3 * FF * 4);
  _Float16* w16l = (_Float16*)carve(128 * 128 * 2);
  _Float16* w16r = (_Float16*)carve(128 * 128 * 2);

  const unsigned ENEG = 0x007FFFFFu;   // fenc(-inf)

  // GraphNorm
  k_set_u32<<<1, 256, 0, stream>>>((unsigned*)sums, 0u, 2 * FF);
  k_col_stats<<<128, 128, 0, stream>>>(x, sums);
  k_gn_finalize<<<1, 128, 0, stream>>>(sums, gw, gb, gms, prm);
  k_gn_apply<<<(NN * FF + 255) / 256, 256, 0, stream>>>(x, prm, h16);

  for (int L = 0; L < 3; ++L) {
    int heads = (L == 2) ? 1 : NHEADS;
    int mout  = heads * CD;            // 128, 128, 32
    // weights -> f16 transposed
    k_wt16<<<(128 * mout + 255) / 256, 256, 0, stream>>>(W[L][0], w16l, mout);
    k_wt16<<<(128 * mout + 255) / 256, 256, 0, stream>>>(W[L][1], w16r, mout);
    // projections (WMMA): NN is exactly 3125 * 16 row tiles
    dim3 gblk(32 * (mout / 16));
    k_gemm_wmma<<<NN / 16, gblk, 0, stream>>>(h16, w16l, Bv[L][0], xl, mout);
    k_gemm_wmma<<<NN / 16, gblk, 0, stream>>>(h16, w16r, Bv[L][1], xr, mout);
    // init segment buffers
    k_set_u32<<<(NN * mout + 255) / 256, 256, 0, stream>>>((unsigned*)agg, 0u, NN * mout);
    k_set_u32<<<(NN * heads + 255) / 256, 256, 0, stream>>>((unsigned*)denom, 0u, NN * heads);
    k_set_u32<<<(NN * heads + 255) / 256, 256, 0, stream>>>(nmax, ENEG, NN * heads);
    // edge phase
    int epb = 256 / (heads * 32);
    int eg  = (NET + epb - 1) / epb;
    k_edge_attn<<<eg, 256, 0, stream>>>(xl, xr, ei, att[L], ebuf, nmax, heads);
    k_edge_exp<<<(NET * heads + 255) / 256, 256, 0, stream>>>(ei, ebuf, nmax, denom, heads);
    k_edge_aggr<<<eg, 256, 0, stream>>>(xl, ei, ebuf, denom, agg, heads);
    if (L < 2)
      k_elu16<<<(NN * HCD + 255) / 256, 256, 0, stream>>>(agg, cbias[L], h16);
  }
  k_final<<<(NN * KOUT + 255) / 256, 256, 0, stream>>>(agg, cbias[2], wlin, blin, out);
}
